// HandwritingTransformer_4475355922977
// MI455X (gfx1250) — matrix operand order repaired and checked by reading, not compiled, not run on hardware
//
#include <hip/hip_runtime.h>
#include <hip/hip_bf16.h>

// ---------------------------------------------------------------------------
// Types for CDNA5 WMMA
// ---------------------------------------------------------------------------
typedef __attribute__((ext_vector_type(16))) __bf16 v16bf;
typedef __attribute__((ext_vector_type(8)))  float  v8f;
typedef __attribute__((ext_vector_type(2)))  __bf16 v2bf;

union FragAB { v16bf v; unsigned short u[16]; };
union FragC  { v8f   v; float          f[8];  };

// hardware bf16 conversion (RTNE fptrunc -> v_cvt_*bf16* on gfx1250)
static __device__ __forceinline__ unsigned short f2bf(float f) {
  union { __bf16 b; unsigned short s; } t;
  t.b = (__bf16)f;
  return t.s;
}
static __device__ __forceinline__ unsigned pk2bf(float lo, float hi) {
  union { v2bf v; unsigned u; } p;
  p.v[0] = (__bf16)lo;
  p.v[1] = (__bf16)hi;
  return p.u;
}
static __device__ __forceinline__ float bf2f(unsigned short h) {
  return __uint_as_float(((unsigned)h) << 16);
}
static __device__ __forceinline__ float bflo(unsigned u) {
  return __uint_as_float(u << 16);
}
static __device__ __forceinline__ float bfhi(unsigned u) {
  return __uint_as_float(u & 0xFFFF0000u);
}
static __device__ __forceinline__ float softplusf(float x) {
  return fmaxf(x, 0.0f) + log1pf(__expf(-fabsf(x)));
}

// ---------------------------------------------------------------------------
// CDNA5 async global->LDS copy (ASYNCcnt path)
// ---------------------------------------------------------------------------
static __device__ __forceinline__ void ldsg_async_b128(unsigned ldsoff,
                                                       const void* g) {
  asm volatile("global_load_async_to_lds_b128 %0, %1, off"
               :: "v"(ldsoff), "v"(g)
               : "memory");
}
static __device__ __forceinline__ void wait_async0() {
  asm volatile("s_wait_asynccnt 0x0" ::: "memory");
}
// Copy n16 chunks of 16B each, global -> LDS, then drain + barrier.
static __device__ __forceinline__ void stage_weights(
    unsigned short* wlds, const unsigned short* wsrc, int n16) {
  const unsigned lbase = (unsigned)(size_t)(void*)wlds;  // LDS aperture low bits
  const char* gsrc = (const char*)wsrc;
  for (int idx = threadIdx.x; idx < n16; idx += 256)
    ldsg_async_b128(lbase + idx * 16, gsrc + idx * 16);
  wait_async0();
  __syncthreads();
}

// ---------------------------------------------------------------------------
// Parameter bundle
// ---------------------------------------------------------------------------
struct BlockW {
  const unsigned short *w_in, *w_out, *w1, *w2;       // bf16 copies (contiguous)
  const float *b_in, *b_out, *b1, *b2;
  const float *ln1g, *ln1b, *ln2g, *ln2b;
};

struct Args {
  const int*            text;        // (8,64)
  const unsigned char*  text_mask;   // (8,64) bool
  const float*          sequences;   // (8,128,4)
  const float*          seqmask;     // (8,128)
  const float *letter_emb, *letter_pos, *cmd_pos, *cmd_w, *cmd_b;
  BlockW blk[3];
  const unsigned short *fa_w_in, *fa_w_out;
  const float *fa_b_in, *fa_b_out;
  const float *out_w, *out_b;
  float *partial;   // ws: per-batch loss sums (8)
  float *y_out;     // d_out: y (8*128*4) then loss at [4096]
};

#define MODE_SPLIT64 0   // QKV / KV split: dst = base + (nb>>6)*12288 + (nb&63)
#define MODE_F32RES  1   // zres[m*64+nb] += val
#define MODE_RELU_BF 2   // outb[m*128+nb] = bf16(relu(val))

// ---------------------------------------------------------------------------
// Tiled bf16 GEMM on WMMA:  OUT(L x N) = A(L x K) @ W(N x K)^T + bias
// A and W both in LDS (row-major). Dual accumulators break the WMMA->WMMA
// D->C dependency chain (avoids mandated hazard NOPs between K-chunks).
// Stores cover full 16-row tiles; rows >= Lv are never consumed downstream.
// ---------------------------------------------------------------------------
static __device__ __forceinline__ void gemm_tiles(
    const unsigned short* A, int lda, int Kdim,
    const unsigned short* __restrict__ Wl, const float* __restrict__ bias,
    int Lt, int Ntiles, int mode,
    unsigned short* outb, float* zres)
{
  const int wave = threadIdx.x >> 5;
  const int lane = threadIdx.x & 31;
  const int lrow = lane & 15;
  const int lhi  = lane >> 4;          // 0 or 1
  const int ntile = Lt * Ntiles;

  for (int t = wave; t < ntile; t += 8) {
    const int rt = t / Ntiles;
    const int ct = t - rt * Ntiles;

    FragC c0, c1;
#pragma unroll
    for (int r = 0; r < 8; r++) { c0.f[r] = 0.0f; c1.f[r] = 0.0f; }

    int half = 0;
    for (int kk = 0; kk < Kdim; kk += 32, half ^= 1) {
      FragAB a, b;
      // A fragment: 16x32 bf16, row = M, contiguous K (k = 16*lhi + e), matching B
      const unsigned short* Ar = A + (rt * 16 + lrow) * lda + kk;
      const int kofs = 16 * lhi;
#pragma unroll
      for (int e = 0; e < 8; e++)  a.u[e] = Ar[e + kofs];
#pragma unroll
      for (int e = 8; e < 16; e++) a.u[e] = Ar[e + kofs];
      // B fragment: 32x16, B[k][n] = W[n][k]; lane holds column n
      const unsigned short* Wr = Wl + (ct * 16 + lrow) * Kdim + kk + 16 * lhi;
#pragma unroll
      for (int e = 0; e < 16; e++) b.u[e] = Wr[e];

      if (half == 0)
        c0.v = __builtin_amdgcn_wmma_f32_16x16x32_bf16(
            false, a.v, false, b.v, (short)0, c0.v, false, false);
      else
        c1.v = __builtin_amdgcn_wmma_f32_16x16x32_bf16(
            false, a.v, false, b.v, (short)0, c1.v, false, false);
    }

    const int nb = ct * 16 + lrow;
    const float bv = bias[nb];
    const int mbase = rt * 16 + 8 * lhi;
    if (mode == MODE_F32RES) {
      float* zr = zres + mbase * 64 + nb;
#pragma unroll
      for (int r = 0; r < 8; r++) zr[r * 64] += c0.f[r] + c1.f[r] + bv;
    } else if (mode == MODE_RELU_BF) {
      unsigned short* dst = outb + mbase * 128 + nb;
#pragma unroll
      for (int r = 0; r < 8; r++)
        dst[r * 128] = f2bf(fmaxf(c0.f[r] + c1.f[r] + bv, 0.0f));
    } else { // MODE_SPLIT64: branch-free q/k/v (or k/v) routing
      unsigned short* dst = outb + (nb >> 6) * 12288 + mbase * 64 + (nb & 63);
#pragma unroll
      for (int r = 0; r < 8; r++) dst[r * 64] = f2bf(c0.f[r] + c1.f[r] + bv);
    }
  }
}

static __device__ __forceinline__ void layernorm_rows(
    float* z, unsigned short* zb, int Lv, const float* g, const float* b)
{
  for (int l = threadIdx.x; l < Lv; l += 256) {
    float* row = z + l * 64;
    float mu = 0.0f;
#pragma unroll 8
    for (int d = 0; d < 64; d++) mu += row[d];
    mu *= (1.0f / 64.0f);
    float var = 0.0f;
#pragma unroll 8
    for (int d = 0; d < 64; d++) { float t = row[d] - mu; var += t * t; }
    var *= (1.0f / 64.0f);
    const float inv = rsqrtf(var + 1e-5f);
    uint4* zo = (uint4*)(zb + l * 64);
#pragma unroll
    for (int q8 = 0; q8 < 8; q8++) {
      float o[8];
#pragma unroll
      for (int j = 0; j < 8; j++) {
        const int d = q8 * 8 + j;
        o[j] = (row[d] - mu) * inv * g[d] + b[d];
        row[d] = o[j];
      }
      uint4 pv;
      pv.x = pk2bf(o[0], o[1]); pv.y = pk2bf(o[2], o[3]);
      pv.z = pk2bf(o[4], o[5]); pv.w = pk2bf(o[6], o[7]);
      zo[q8] = pv;
    }
  }
}

// One post-norm transformer block, in place on z (f32) / zb (bf16 mirror).
// Weights are staged to LDS (wlds) via async copy first.
static __device__ __forceinline__ void run_block(
    float* z, unsigned short* zb, unsigned short* sbuf, unsigned short* wlds,
    const unsigned char* keyok, int Lv, const BlockW& w)
{
  unsigned short* qb = sbuf;
  unsigned short* kb = sbuf + 12288;
  unsigned short* vb = sbuf + 24576;
  unsigned short* hb = sbuf;           // FFN hidden aliases q|k (dead then)
  const int Lt = (Lv + 15) >> 4;

  // stage w_in|w_out|w1|w2 (contiguous 32768 bf16 = 64KB) into LDS
  stage_weights(wlds, w.w_in, 4096);
  const unsigned short* w_in  = wlds;
  const unsigned short* w_out = wlds + 12288;
  const unsigned short* w1    = wlds + 16384;
  const unsigned short* w2    = wlds + 24576;

  // q,k,v projections (fused N=192 GEMM, branch-free split store)
  gemm_tiles(zb, 64, 64, w_in, w.b_in, Lt, 12, MODE_SPLIT64, sbuf, z);
  __syncthreads();

  // attention: per (query,head) online softmax over keys (dh = 8)
  const float scale = 0.35355339059327373f;
  for (int p = threadIdx.x; p < Lv * 8; p += 256) {
    const int l = p >> 3, h = p & 7;
    const uint4 qu = *(const uint4*)(qb + l * 64 + h * 8);
    float qd[8];
    qd[0] = bflo(qu.x); qd[1] = bfhi(qu.x); qd[2] = bflo(qu.y); qd[3] = bfhi(qu.y);
    qd[4] = bflo(qu.z); qd[5] = bfhi(qu.z); qd[6] = bflo(qu.w); qd[7] = bfhi(qu.w);
    float m = -1e30f, s = 0.0f, acc[8];
#pragma unroll
    for (int d = 0; d < 8; d++) acc[d] = 0.0f;
    for (int kk = 0; kk < Lv; kk++) {
      if (keyok && !keyok[kk]) continue;
      const uint4 ku = *(const uint4*)(kb + kk * 64 + h * 8);
      float sc = qd[0] * bflo(ku.x) + qd[1] * bfhi(ku.x)
               + qd[2] * bflo(ku.y) + qd[3] * bfhi(ku.y)
               + qd[4] * bflo(ku.z) + qd[5] * bfhi(ku.z)
               + qd[6] * bflo(ku.w) + qd[7] * bfhi(ku.w);
      sc *= scale;
      const float mn = fmaxf(m, sc);
      const float cc = __expf(m - mn), pe = __expf(sc - mn);
      s = s * cc + pe;
      const uint4 vu = *(const uint4*)(vb + kk * 64 + h * 8);
      acc[0] = acc[0] * cc + pe * bflo(vu.x);
      acc[1] = acc[1] * cc + pe * bfhi(vu.x);
      acc[2] = acc[2] * cc + pe * bflo(vu.y);
      acc[3] = acc[3] * cc + pe * bfhi(vu.y);
      acc[4] = acc[4] * cc + pe * bflo(vu.z);
      acc[5] = acc[5] * cc + pe * bfhi(vu.z);
      acc[6] = acc[6] * cc + pe * bflo(vu.w);
      acc[7] = acc[7] * cc + pe * bfhi(vu.w);
      m = mn;
    }
    const float inv = 1.0f / s;
    uint4 ou;
    ou.x = pk2bf(acc[0] * inv, acc[1] * inv);
    ou.y = pk2bf(acc[2] * inv, acc[3] * inv);
    ou.z = pk2bf(acc[4] * inv, acc[5] * inv);
    ou.w = pk2bf(acc[6] * inv, acc[7] * inv);
    *(uint4*)(zb + l * 64 + h * 8) = ou;
  }
  __syncthreads();

  // out projection + residual:  z = x + attn_out
  gemm_tiles(zb, 64, 64, w_out, w.b_out, Lt, 4, MODE_F32RES, sbuf, z);
  __syncthreads();
  layernorm_rows(z, zb, Lv, w.ln1g, w.ln1b);
  __syncthreads();

  // FFN (hb aliases q|k space; those are dead now)
  gemm_tiles(zb, 64, 64, w1, w.b1, Lt, 8, MODE_RELU_BF, hb, z);
  __syncthreads();
  gemm_tiles(hb, 128, 128, w2, w.b2, Lt, 4, MODE_F32RES, sbuf, z);
  __syncthreads();
  layernorm_rows(z, zb, Lv, w.ln2g, w.ln2b);
  __syncthreads();
}

// ---------------------------------------------------------------------------
// Kernels
// ---------------------------------------------------------------------------
__global__ void cvt_bf16(const float* __restrict__ src,
                         unsigned short* __restrict__ dst, int n) {
  int i = blockIdx.x * 256 + threadIdx.x;
  if (i < n) dst[i] = f2bf(src[i]);
}

__global__ __launch_bounds__(256) void hw_main(Args A) {
  __shared__ __align__(16) float          latl[192 * 64];     // persistent latent
  __shared__ __align__(16) float          z[192 * 64];        // working activations
  __shared__ __align__(16) unsigned short zb[192 * 64];       // bf16 mirror of z
  __shared__ __align__(16) unsigned short sbuf[3 * 192 * 64]; // q | k | v (hb alias)
  __shared__ __align__(16) unsigned short wlds[32768];        // staged weights 64KB
  __shared__ unsigned char keyok[192];
  __shared__ float         o64[64], oo[64], pp6[8], ycur[4];
  __shared__ float         lossacc;

  unsigned short* kb = sbuf + 12288;
  unsigned short* vb = sbuf + 24576;

  const int b    = blockIdx.x;
  const int tid  = threadIdx.x;
  const int wave = tid >> 5, lane = tid & 31;

  if (tid == 0) lossacc = 0.0f;
  if (tid < 4)  ycur[tid] = 0.0f;

  // ---- block1 input: letter_emb[text] + letter_pos, rows 0..63 ----
  for (int idx = tid; idx < 64 * 64; idx += 256) {
    const int t = idx >> 6, d = idx & 63;
    const int ch = A.text[b * 64 + t];
    const float xv = A.letter_emb[ch * 64 + d] + A.letter_pos[t * 64 + d];
    z[t * 64 + d]  = xv;
    zb[t * 64 + d] = f2bf(xv);
  }
  for (int t = tid; t < 192; t += 256)
    keyok[t] = (t < 64) ? (A.text_mask[b * 64 + t] ? 0 : 1) : 0;
  __syncthreads();

  run_block(z, zb, sbuf, wlds, keyok, 64, A.blk[0]);

  // latent0: rows 0..63 = block1 output, rows 64..191 = 0
  for (int idx = tid; idx < 192 * 64; idx += 256)
    latl[idx] = (idx < 64 * 64) ? z[idx] : 0.0f;
  __syncthreads();

  // ---- scan over 128 steps ----
#pragma unroll 1
  for (int i = 0; i < 128; i++) {
    const int Lv = 64 + i;         // valid key/query rows this step
    const int Lt = (Lv + 15) >> 4;

    // latent[:, 64+i] += (i>0 ? prev_y @ cmd_w^T + cmd_b : 0) + cmd_pos[i]
    if (tid < 64) {
      float add = 0.0f;
      if (i > 0) {
#pragma unroll
        for (int j = 0; j < 4; j++) add += ycur[j] * A.cmd_w[tid * 4 + j];
        add += A.cmd_b[tid];
      }
      latl[(64 + i) * 64 + tid] += add + A.cmd_pos[i * 64 + tid];
    }
    // refresh working copy from latent (vectorized)
    for (int idx = tid; idx < Lv * 16; idx += 256) {
      const float4 v = ((const float4*)latl)[idx];
      ((float4*)z)[idx] = v;
      unsigned* zo = (unsigned*)zb + idx * 2;
      zo[0] = pk2bf(v.x, v.y);
      zo[1] = pk2bf(v.z, v.w);
    }
    __syncthreads();

    run_block(z, zb, sbuf, wlds, (const unsigned char*)nullptr, Lv, A.blk[1]);
    run_block(z, zb, sbuf, wlds, (const unsigned char*)nullptr, Lv, A.blk[2]);

    // ---- final attention: q = fa.b_in[:64] (zero query input) ----
    // stage fa w_in|w_out (16384 bf16 = 32KB) into LDS
    stage_weights(wlds, A.fa_w_in, 2048);
    gemm_tiles(zb, 64, 64, wlds + 64 * 64, A.fa_b_in + 64,
               Lt, 8, MODE_SPLIT64, kb, z);
    __syncthreads();

    {
      const int h = wave;                  // 8 waves == 8 heads
      float qd[8];
#pragma unroll
      for (int d = 0; d < 8; d++) qd[d] = A.fa_b_in[h * 8 + d];
      float m = -1e30f, s = 0.0f, acc[8];
#pragma unroll
      for (int d = 0; d < 8; d++) acc[d] = 0.0f;
      for (int kk = lane; kk < Lv; kk += 32) {
        const uint4 ku = *(const uint4*)(kb + kk * 64 + h * 8);
        float sc = qd[0] * bflo(ku.x) + qd[1] * bfhi(ku.x)
                 + qd[2] * bflo(ku.y) + qd[3] * bfhi(ku.y)
                 + qd[4] * bflo(ku.z) + qd[5] * bfhi(ku.z)
                 + qd[6] * bflo(ku.w) + qd[7] * bfhi(ku.w);
        sc *= 0.35355339059327373f;
        const float mn = fmaxf(m, sc);
        const float cc = __expf(m - mn), pe = __expf(sc - mn);
        s = s * cc + pe;
        const uint4 vu = *(const uint4*)(vb + kk * 64 + h * 8);
        acc[0] = acc[0] * cc + pe * bflo(vu.x);
        acc[1] = acc[1] * cc + pe * bfhi(vu.x);
        acc[2] = acc[2] * cc + pe * bflo(vu.y);
        acc[3] = acc[3] * cc + pe * bfhi(vu.y);
        acc[4] = acc[4] * cc + pe * bflo(vu.z);
        acc[5] = acc[5] * cc + pe * bfhi(vu.z);
        acc[6] = acc[6] * cc + pe * bflo(vu.w);
        acc[7] = acc[7] * cc + pe * bfhi(vu.w);
        m = mn;
      }
      // wave-level softmax-state combine
      for (int off = 16; off > 0; off >>= 1) {
        const float m2 = __shfl_xor(m, off, 32);
        const float s2 = __shfl_xor(s, off, 32);
        float a2[8];
#pragma unroll
        for (int d = 0; d < 8; d++) a2[d] = __shfl_xor(acc[d], off, 32);
        const float mn = fmaxf(m, m2);
        const float c1 = __expf(m - mn), c2 = __expf(m2 - mn);
        s = s * c1 + s2 * c2;
#pragma unroll
        for (int d = 0; d < 8; d++) acc[d] = acc[d] * c1 + a2[d] * c2;
        m = mn;
      }
      if (lane == 0) {
        const float inv = 1.0f / s;
#pragma unroll
        for (int d = 0; d < 8; d++) o64[h * 8 + d] = acc[d] * inv;
      }
    }
    __syncthreads();

    // zq = o @ fa.w_out^T + fa.b_out   (64x64, from staged LDS weights)
    if (tid < 64) {
      float r = A.fa_b_out[tid];
      const unsigned short* wr = wlds + 12288 + tid * 64;
#pragma unroll 8
      for (int d = 0; d < 64; d++) r += o64[d] * bf2f(wr[d]);
      oo[tid] = r;
    }
    __syncthreads();
    // pp = zq @ out_w^T + out_b  (6 outputs)
    if (tid < 6) {
      float r = A.out_b[tid];
      const float* wr = A.out_w + tid * 64;
#pragma unroll 8
      for (int d = 0; d < 64; d++) r += oo[d] * wr[d];
      pp6[tid] = r;
    }
    __syncthreads();

    // step tail: y update + log-prob
    if (tid == 0) {
      const float mu0 = pp6[0], mu1 = pp6[1];
      const float lv0 = pp6[2], lv1 = pp6[3];
      const float ls = pp6[4], le = pp6[5];
      const int ip = (i + 127) & 127;      // (i-1) mod 128
      const float* sp = A.sequences + (b * 128 + ip) * 4;
      const float stroke = (ls > 0.0f) ? 1.0f : 0.0f;
      const float ep     = (le > 0.0f) ? 1.0f : 0.0f;
      const float y0 = sp[0] + mu0, y1 = sp[1] + mu1;
      const float y2 = sp[2] + stroke, y3 = sp[3] + ep;
      float* yo = A.y_out + (b * 128 + i) * 4;
      yo[0] = y0; yo[1] = y1; yo[2] = y2; yo[3] = y3;
      ycur[0] = y0; ycur[1] = y1; ycur[2] = y2; ycur[3] = y3;
      const float d0 = sp[0] - mu0, d1 = sp[1] - mu1;
      float lp = -0.5f * (d0 * d0 * __expf(-lv0) + lv0 +
                          d1 * d1 * __expf(-lv1) + lv1)
                 - 1.8378770664093453f;
      lp += -(sp[2] * softplusf(-ls) + (1.0f - sp[2]) * softplusf(ls));
      lp += -(sp[3] * softplusf(-le) + (1.0f - sp[3]) * softplusf(le));
      lossacc += lp * A.seqmask[b * 128 + i];
    }
    __syncthreads();
  }

  if (tid == 0) A.partial[b] = lossacc;
}

__global__ void hw_finalize(const float* __restrict__ partial,
                            const float* __restrict__ mask,
                            float* __restrict__ out) {
  __shared__ float red[256];
  const int tid = threadIdx.x;
  float s = 0.0f;
  for (int idx = tid; idx < 1024; idx += 256) s += mask[idx];
  red[tid] = s;
  __syncthreads();
  for (int off = 128; off > 0; off >>= 1) {
    if (tid < off) red[tid] += red[tid + off];
    __syncthreads();
  }
  if (tid == 0) {
    float lp = 0.0f;
    for (int bb = 0; bb < 8; bb++) lp += partial[bb];
    out[4096] = lp / red[0];
  }
}

// ---------------------------------------------------------------------------
// Host entry
// ---------------------------------------------------------------------------
extern "C" void kernel_launch(void* const* d_in, const int* in_sizes, int n_in,
                              void* d_out, int out_size, void* d_ws, size_t ws_size,
                              hipStream_t stream) {
  (void)in_sizes; (void)n_in; (void)out_size; (void)ws_size;

  unsigned short* bfw = (unsigned short*)d_ws;
  static const int widx[14] = {9, 11, 13, 15, 21, 23, 25, 27, 33, 35, 37, 39, 45, 47};
  static const int wn[14]   = {12288, 4096, 8192, 8192, 12288, 4096, 8192, 8192,
                               12288, 4096, 8192, 8192, 12288, 4096};
  size_t offs[14], acc = 0;
  for (int k = 0; k < 14; k++) { offs[k] = acc; acc += (size_t)wn[k]; }
  for (int k = 0; k < 14; k++) {
    const int nb = (wn[k] + 255) / 256;
    cvt_bf16<<<nb, 256, 0, stream>>>((const float*)d_in[widx[k]], bfw + offs[k], wn[k]);
  }
  float* partial = (float*)((char*)d_ws + acc * sizeof(unsigned short));

  Args A;
  A.text       = (const int*)d_in[0];
  A.text_mask  = (const unsigned char*)d_in[1];
  A.sequences  = (const float*)d_in[2];
  A.seqmask    = (const float*)d_in[3];
  A.letter_emb = (const float*)d_in[4];
  A.letter_pos = (const float*)d_in[5];
  A.cmd_pos    = (const float*)d_in[6];
  A.cmd_w      = (const float*)d_in[7];
  A.cmd_b      = (const float*)d_in[8];
  for (int bl = 0; bl < 3; bl++) {
    const int o = 9 + bl * 12;
    A.blk[bl].w_in  = bfw + offs[bl * 4 + 0];
    A.blk[bl].w_out = bfw + offs[bl * 4 + 1];
    A.blk[bl].w1    = bfw + offs[bl * 4 + 2];
    A.blk[bl].w2    = bfw + offs[bl * 4 + 3];
    A.blk[bl].b_in  = (const float*)d_in[o + 1];
    A.blk[bl].b_out = (const float*)d_in[o + 3];
    A.blk[bl].b1    = (const float*)d_in[o + 5];
    A.blk[bl].b2    = (const float*)d_in[o + 7];
    A.blk[bl].ln1g  = (const float*)d_in[o + 8];
    A.blk[bl].ln1b  = (const float*)d_in[o + 9];
    A.blk[bl].ln2g  = (const float*)d_in[o + 10];
    A.blk[bl].ln2b  = (const float*)d_in[o + 11];
  }
  A.fa_w_in  = bfw + offs[12];
  A.fa_w_out = bfw + offs[13];
  A.fa_b_in  = (const float*)d_in[46];
  A.fa_b_out = (const float*)d_in[48];
  A.out_w    = (const float*)d_in[49];
  A.out_b    = (const float*)d_in[50];
  A.partial  = partial;
  A.y_out    = (float*)d_out;

  hw_main<<<8, 256, 0, stream>>>(A);
  hw_finalize<<<1, 256, 0, stream>>>(partial, (const float*)d_in[3], (float*)d_out);
}
